// SelfAttention_6296422056247
// MI455X (gfx1250) — matrix operand order repaired and checked by reading, not compiled, not run on hardware
//
#include <hip/hip_runtime.h>
#include <math.h>

// ---------------------------------------------------------------------------
// Problem constants (match the reference)
// ---------------------------------------------------------------------------
constexpr int B_ = 16, S_ = 2048, D_ = 512, H_ = 512;

// GEMM tiling: 256 threads = 8 wave32 waves, block tile 128x64, K-step 32.
// Each wave computes a 32x32 tile = 2x2 v_wmma_f32_16x16x32_bf16 accumulators.
constexpr int BM = 128, BN = 64, BK = 32;

typedef __attribute__((ext_vector_type(16))) __bf16 v16bf;
typedef __attribute__((ext_vector_type(8)))  __bf16 v8bf;
typedef __attribute__((ext_vector_type(8)))  float  v8f;

#define WMMA_BF16(a, b, c) \
    __builtin_amdgcn_wmma_f32_16x16x32_bf16(false, (a), false, (b), (short)0, (c), false, false)

// ---------------------------------------------------------------------------
// CDNA5 async global->LDS copies (ASYNCcnt path). The builtin takes typed
// int4 pointers: (int4 addrspace(1)* gsrc, int4 addrspace(3)* ldst, imm, imm).
// Guarded so a missing builtin cannot break the build.
// ---------------------------------------------------------------------------
#if defined(__has_builtin)
#if __has_builtin(__builtin_amdgcn_global_load_async_to_lds_b128) && \
    __has_builtin(__builtin_amdgcn_s_wait_asynccnt)
#define USE_ASYNC_LDS 1
#endif
#endif
#ifndef USE_ASYNC_LDS
#define USE_ASYNC_LDS 0
#endif

#if USE_ASYNC_LDS
typedef int v4i_ __attribute__((vector_size(16)));
typedef __attribute__((address_space(1))) v4i_ gm_v4i;
typedef __attribute__((address_space(3))) v4i_ lds_v4i;
__device__ inline void async_cp_b128(const __bf16* gsrc, __bf16* lds_dst) {
    __builtin_amdgcn_global_load_async_to_lds_b128(
        (gm_v4i*)gsrc, (lds_v4i*)lds_dst, 0, 0);
}
__device__ inline void async_wait0() { __builtin_amdgcn_s_wait_asynccnt(0); }
#else
__device__ inline void async_wait0() {}
#endif

// ---------------------------------------------------------------------------
// Fragment-major LDS layout.
// A tile (BM x BK) is 8 sub-tiles of 16 rows; B tile (BK x BN) is 4 sub-tiles
// of 16 cols. Each sub-tile is stored as [lane][16 bf16] (32 B contiguous per
// lane) following the CDNA5 WMMA VGPR layouts (CDNA5 ISA):
//   A 16x32: lanes 0-15 hold M=lane, K in {0..7,16..23}; lanes 16-31 get K+8.
//   B 32x16: lanes 0-15 hold N=lane, K in {0..7,16..23}; lanes 16-31 get K+8.
//   C 16x16: vgpr r, lane l -> (M = r + 8*(l>=16), N = l&15).
// Fragment load = one aligned 32 B LDS read (2x ds_load_b128).
// ---------------------------------------------------------------------------
__device__ inline v16bf frag_load(const __bf16* p) { return *(const v16bf*)p; }

__device__ inline __bf16* afrag(__bf16* Af, int mbase16, int lane) {
    return Af + (((mbase16 >> 4) * 32 + lane) << 4);
}
__device__ inline __bf16* bfrag(__bf16* Bf, int nbase16, int lane) {
    return Bf + (((nbase16 >> 4) * 32 + lane) << 4);
}

// A-tile writer slots: thread (r = tid>>1 in 0..127, ch = tid&1 selects
// K-half 0..15 / 16..31). Global src = row r, 16 consecutive K. The first 8
// K's land in lane (r&15), the next 8 in lane (r&15)+16, both at element
// offset ch*8 -> two contiguous 16 B LDS stores (async_cp_b128-compatible).
struct AWriter {
    __bf16 *s0, *s1;
    __device__ AWriter(__bf16* Af, int tid) {
        int r = tid >> 1, ch = tid & 1;
        int mt = r >> 4, mr = r & 15, eb = ch * 8;
        s0 = Af + ((mt * 32 + mr) << 4) + eb;
        s1 = Af + ((mt * 32 + mr + 16) << 4) + eb;
    }
};

// B-tile writer slot: thread (n = tid&63, kb = (tid>>6)*8 in {0,8,16,24}).
// 8 consecutive K's for one n -> one contiguous 16 B LDS store.
struct BWriter {
    __bf16* d;
    __device__ BWriter(__bf16* Bf, int tid) {
        int n = tid & 63, kb = (tid >> 6) * 8;
        int nt = n >> 4, nr = n & 15;
        d = Bf + (((nt * 32 + nr + 16 * ((kb >> 3) & 1)) << 4) + ((kb >> 4) * 8));
    }
};

// ---------------------------------------------------------------------------
// Kernel 1: fused triple projection  O_z = audio @ W_z + b_z  (bf16 out)
//   z=0: phi [B*S,H]; z=1: theta [B*S,H]; z=2: g written TRANSPOSED as
//   gT[b][h][t] (packed v8bf stores) so the attn kernel's B tile is a pure
//   contiguous global->LDS copy.
// ---------------------------------------------------------------------------
__global__ __launch_bounds__(256)
void proj3_kernel(const float* __restrict__ A,
                  const float* __restrict__ W0, const float* __restrict__ bb0,
                  const float* __restrict__ W1, const float* __restrict__ bb1,
                  const float* __restrict__ W2, const float* __restrict__ bb2,
                  __bf16* __restrict__ O0, __bf16* __restrict__ O1,
                  __bf16* __restrict__ GT)
{
    const float* W; const float* bias;
    if (blockIdx.z == 0)      { W = W0; bias = bb0; }
    else if (blockIdx.z == 1) { W = W1; bias = bb1; }
    else                      { W = W2; bias = bb2; }

    __shared__ __align__(32) __bf16 Af[(BM / 16) * 32 * 16];   // 8 KB
    __shared__ __align__(32) __bf16 Bf[(BN / 16) * 32 * 16];   // 4 KB

    const int tid  = threadIdx.x;
    const int lane = tid & 31, wave = tid >> 5;
    const int wm = (wave >> 1) * 32, wn = (wave & 1) * 32;
    const int m0 = blockIdx.x * BM;
    const int n0 = blockIdx.y * BN;

    AWriter aw(Af, tid);
    BWriter bw(Bf, tid);
    const int ar = tid >> 1, ach = (tid & 1) * 16;
    const int bn = tid & 63, bkb = (tid >> 6) * 8;

    v8f acc[2][2] = {};

    for (int k0 = 0; k0 < D_; k0 += BK) {
        { // A tile: audio fp32 -> bf16, two 16 B lane-slot stores
            const float* src = A + (size_t)(m0 + ar) * D_ + k0 + ach;
#pragma unroll
            for (int e = 0; e < 8; ++e) {
                aw.s0[e] = (__bf16)src[e];
                aw.s1[e] = (__bf16)src[8 + e];
            }
        }
        { // W tile: 8 K's (stride H_) for one n, fp32 -> bf16, 16 B store
            const float* src = W + (size_t)(k0 + bkb) * H_ + n0 + bn;
#pragma unroll
            for (int e = 0; e < 8; ++e) bw.d[e] = (__bf16)src[(size_t)e * H_];
        }
        __syncthreads();

        v16bf a0 = frag_load(afrag(Af, wm, lane));
        v16bf a1 = frag_load(afrag(Af, wm + 16, lane));
        v16bf f0 = frag_load(bfrag(Bf, wn, lane));
        v16bf f1 = frag_load(bfrag(Bf, wn + 16, lane));
        acc[0][0] = WMMA_BF16(a0, f0, acc[0][0]);
        acc[0][1] = WMMA_BF16(a0, f1, acc[0][1]);
        acc[1][0] = WMMA_BF16(a1, f0, acc[1][0]);
        acc[1][1] = WMMA_BF16(a1, f1, acc[1][1]);
        __syncthreads();
    }

    const int nl = lane & 15, mh = (lane >> 4) * 8;
    if (blockIdx.z < 2) {
        __bf16* O = (blockIdx.z == 0) ? O0 : O1;
#pragma unroll
        for (int fm = 0; fm < 2; ++fm)
#pragma unroll
            for (int fn = 0; fn < 2; ++fn) {
                int gn = n0 + wn + fn * 16 + nl;
                float bv = bias[gn];
#pragma unroll
                for (int r = 0; r < 8; ++r) {
                    int gm = m0 + wm + fm * 16 + mh + r;
                    O[(size_t)gm * H_ + gn] = (__bf16)(acc[fm][fn][r] + bv);
                }
            }
    } else {
        // g transposed: gT[b][h][t]; per lane the 8 rows are consecutive t
        // -> pack into one 16 B store.
#pragma unroll
        for (int fm = 0; fm < 2; ++fm)
#pragma unroll
            for (int fn = 0; fn < 2; ++fn) {
                int gn = n0 + wn + fn * 16 + nl;
                float bv = bias[gn];
                v8bf pk;
#pragma unroll
                for (int r = 0; r < 8; ++r) pk[r] = (__bf16)(acc[fm][fn][r] + bv);
                int mbase = m0 + wm + fm * 16 + mh;       // global token index
                int bb = mbase >> 11;                      // / S_
                int tt = mbase & (S_ - 1);
                *(v8bf*)(GT + ((size_t)bb * H_ + gn) * S_ + tt) = pk;
            }
    }
}

// ---------------------------------------------------------------------------
// Kernel 2: raw relation  R = relu(scale * phi @ theta^T)  per batch, fp32 out
//   Both tiles are bf16 16B/lane copies -> async global->LDS when available.
// ---------------------------------------------------------------------------
__global__ __launch_bounds__(256)
void rel_raw_kernel(const __bf16* __restrict__ Phi, const __bf16* __restrict__ Theta,
                    float* __restrict__ R, float scale)
{
    const int b = blockIdx.z;
    const __bf16* phi   = Phi   + (size_t)b * S_ * H_;
    const __bf16* theta = Theta + (size_t)b * S_ * H_;
    float* Rb = R + (size_t)b * S_ * S_;

    __shared__ __align__(32) __bf16 Af[(BM / 16) * 32 * 16];
    __shared__ __align__(32) __bf16 Bf[(BN / 16) * 32 * 16];

    const int tid  = threadIdx.x;
    const int lane = tid & 31, wave = tid >> 5;
    const int wm = (wave >> 1) * 32, wn = (wave & 1) * 32;
    const int m0 = blockIdx.x * BM;
    const int n0 = blockIdx.y * BN;

    AWriter aw(Af, tid);
    BWriter bw(Bf, tid);
    const int ar = tid >> 1, ach = (tid & 1) * 16;
    const int bn = tid & 63, bkb = (tid >> 6) * 8;

    v8f acc[2][2] = {};

    for (int k0 = 0; k0 < H_; k0 += BK) {
        const __bf16* asrc = phi + (size_t)(m0 + ar) * H_ + k0 + ach;
        const __bf16* bsrc = theta + (size_t)(n0 + bn) * H_ + k0 + bkb;
#if USE_ASYNC_LDS
        async_cp_b128(asrc, aw.s0);
        async_cp_b128(asrc + 8, aw.s1);
        async_cp_b128(bsrc, bw.d);
        async_wait0();
#else
#pragma unroll
        for (int e = 0; e < 8; ++e) { aw.s0[e] = asrc[e]; aw.s1[e] = asrc[8 + e]; }
#pragma unroll
        for (int e = 0; e < 8; ++e) bw.d[e] = bsrc[e];
#endif
        __syncthreads();

        v16bf a0 = frag_load(afrag(Af, wm, lane));
        v16bf a1 = frag_load(afrag(Af, wm + 16, lane));
        v16bf f0 = frag_load(bfrag(Bf, wn, lane));
        v16bf f1 = frag_load(bfrag(Bf, wn + 16, lane));
        acc[0][0] = WMMA_BF16(a0, f0, acc[0][0]);
        acc[0][1] = WMMA_BF16(a0, f1, acc[0][1]);
        acc[1][0] = WMMA_BF16(a1, f0, acc[1][0]);
        acc[1][1] = WMMA_BF16(a1, f1, acc[1][1]);
        __syncthreads();
    }

    const int nl = lane & 15, mh = (lane >> 4) * 8;
#pragma unroll
    for (int fm = 0; fm < 2; ++fm)
#pragma unroll
        for (int fn = 0; fn < 2; ++fn) {
            int gn = n0 + wn + fn * 16 + nl;
#pragma unroll
            for (int r = 0; r < 8; ++r) {
                int gm = m0 + wm + fm * 16 + mh + r;
                Rb[(size_t)gm * S_ + gn] = fmaxf(acc[fm][fn][r] * scale, 0.0f);
            }
        }
}

// ---------------------------------------------------------------------------
// Kernel 3/4: deterministic row / column sums of raw R (no atomics)
// ---------------------------------------------------------------------------
__global__ __launch_bounds__(256)
void rowsum_kernel(const float* __restrict__ R, float* __restrict__ rowsum)
{
    size_t row = blockIdx.x;                 // 0 .. B*S-1
    const float* p = R + row * (size_t)S_;
    float s = 0.0f;
    for (int i = threadIdx.x; i < S_; i += 256) s += p[i];
    __shared__ float red[256];
    red[threadIdx.x] = s;
    __syncthreads();
    for (int off = 128; off > 0; off >>= 1) {
        if (threadIdx.x < off) red[threadIdx.x] += red[threadIdx.x + off];
        __syncthreads();
    }
    if (threadIdx.x == 0) rowsum[row] = red[0];
}

__global__ __launch_bounds__(256)
void colsum_kernel(const float* __restrict__ R, float* __restrict__ colsum)
{
    int b = blockIdx.y;
    int t = blockIdx.x * 256 + threadIdx.x;
    const float* p = R + (size_t)b * S_ * S_ + t;
    float s = 0.0f;
    for (int ss = 0; ss < S_; ++ss) s += p[(size_t)ss * S_];
    colsum[(size_t)b * S_ + t] = s;
}

// ---------------------------------------------------------------------------
// Kernel 5: in-place symmetrize + row-normalize on tile pairs (i,j)/(j,i).
//   rel[s,t] = 0.5*(R[s,t]+R[t,s]) / (0.5*(rowsum[s]+colsum[s]) + 1e-8)
// Each unordered tile pair is owned by exactly one block -> no races.
// ---------------------------------------------------------------------------
constexpr int TS = 64;
__global__ __launch_bounds__(256)
void symnorm_kernel(float* __restrict__ R,
                    const float* __restrict__ rowsum, const float* __restrict__ colsum)
{
    const int i = blockIdx.x, j = blockIdx.y, b = blockIdx.z;
    if (j < i) return;                       // upper-triangular tile pairs only
    const int i0 = i * TS, j0 = j * TS;
    float* Rb = R + (size_t)b * S_ * S_;
    const float* rs = rowsum + (size_t)b * S_;
    const float* cs = colsum + (size_t)b * S_;

    __shared__ float tB[TS][TS + 1];         // tile (j,i)
    __shared__ float tSym[TS][TS + 1];       // symmetrized values
    const int tid = threadIdx.x;

#pragma unroll
    for (int e = 0; e < 16; ++e) {
        int idx = tid + 256 * e, r = idx >> 6, c = idx & 63;
        tB[r][c] = Rb[(size_t)(j0 + r) * S_ + i0 + c];
    }
    __syncthreads();

#pragma unroll
    for (int e = 0; e < 16; ++e) {
        int idx = tid + 256 * e, r = idx >> 6, c = idx & 63;
        int s = i0 + r, t = j0 + c;
        float a   = Rb[(size_t)s * S_ + t];
        float sym = 0.5f * (a + tB[c][r]);
        tSym[r][c] = sym;
        float den = 0.5f * (rs[s] + cs[s]) + 1e-8f;
        Rb[(size_t)s * S_ + t] = sym / den;
    }
    __syncthreads();

#pragma unroll
    for (int e = 0; e < 16; ++e) {           // mirrored tile, coalesced writes
        int idx = tid + 256 * e, r = idx >> 6, c = idx & 63;
        int t = j0 + r, s = i0 + c;
        float den = 0.5f * (rs[t] + cs[t]) + 1e-8f;
        Rb[(size_t)t * S_ + s] = tSym[c][r] / den;
    }
}

// ---------------------------------------------------------------------------
// Kernel 6: a_att = rel @ g + audio  per batch.
//   A tile: rel fp32 -> bf16 (VGPR conversion). B tile: gT[b][h][t] is a
//   contiguous bf16 copy -> async global->LDS when available.
// ---------------------------------------------------------------------------
__global__ __launch_bounds__(256)
void attn_kernel(const float* __restrict__ R, const __bf16* __restrict__ GT,
                 const float* __restrict__ audio, float* __restrict__ out)
{
    const int b = blockIdx.z;
    const float*  Rb  = R     + (size_t)b * S_ * S_;
    const __bf16* GTb = GT    + (size_t)b * H_ * S_;   // [h][t]
    const float*  Ab  = audio + (size_t)b * S_ * H_;
    float*        Ob  = out   + (size_t)b * S_ * H_;

    __shared__ __align__(32) __bf16 Af[(BM / 16) * 32 * 16];
    __shared__ __align__(32) __bf16 Bf[(BN / 16) * 32 * 16];

    const int tid  = threadIdx.x;
    const int lane = tid & 31, wave = tid >> 5;
    const int wm = (wave >> 1) * 32, wn = (wave & 1) * 32;
    const int m0 = blockIdx.x * BM;
    const int n0 = blockIdx.y * BN;

    AWriter aw(Af, tid);
    BWriter bw(Bf, tid);
    const int ar = tid >> 1, ach = (tid & 1) * 16;
    const int bn = tid & 63, bkb = (tid >> 6) * 8;

    v8f acc[2][2] = {};

    for (int k0 = 0; k0 < S_; k0 += BK) {     // K dim = t (2048)
        { // A tile: rel fp32 -> bf16
            const float* src = Rb + (size_t)(m0 + ar) * S_ + k0 + ach;
#pragma unroll
            for (int e = 0; e < 8; ++e) {
                aw.s0[e] = (__bf16)src[e];
                aw.s1[e] = (__bf16)src[8 + e];
            }
        }
        { // B tile: gT row h = n0+bn, 8 consecutive t
            const __bf16* src = GTb + (size_t)(n0 + bn) * S_ + k0 + bkb;
#if USE_ASYNC_LDS
            async_cp_b128(src, bw.d);
            async_wait0();
#else
#pragma unroll
            for (int e = 0; e < 8; ++e) bw.d[e] = src[e];
#endif
        }
        __syncthreads();

        v16bf a0 = frag_load(afrag(Af, wm, lane));
        v16bf a1 = frag_load(afrag(Af, wm + 16, lane));
        v16bf f0 = frag_load(bfrag(Bf, wn, lane));
        v16bf f1 = frag_load(bfrag(Bf, wn + 16, lane));
        acc[0][0] = WMMA_BF16(a0, f0, acc[0][0]);
        acc[0][1] = WMMA_BF16(a0, f1, acc[0][1]);
        acc[1][0] = WMMA_BF16(a1, f0, acc[1][0]);
        acc[1][1] = WMMA_BF16(a1, f1, acc[1][1]);
        __syncthreads();
    }

    const int nl = lane & 15, mh = (lane >> 4) * 8;
#pragma unroll
    for (int fm = 0; fm < 2; ++fm)
#pragma unroll
        for (int fn = 0; fn < 2; ++fn) {
            int gn = n0 + wn + fn * 16 + nl;
#pragma unroll
            for (int r = 0; r < 8; ++r) {
                int gm = m0 + wm + fm * 16 + mh + r;
                size_t off = (size_t)gm * H_ + gn;
                Ob[off] = acc[fm][fn][r] + Ab[off];   // residual add
            }
        }
}

// ---------------------------------------------------------------------------
// Host launch
// ---------------------------------------------------------------------------
extern "C" void kernel_launch(void* const* d_in, const int* in_sizes, int n_in,
                              void* d_out, int out_size, void* d_ws, size_t ws_size,
                              hipStream_t stream)
{
    const float* audio = (const float*)d_in[0];
    const float* Wphi  = (const float*)d_in[1];
    const float* bphi  = (const float*)d_in[2];
    const float* Wth   = (const float*)d_in[3];
    const float* bth   = (const float*)d_in[4];
    const float* Wg    = (const float*)d_in[5];
    const float* bg    = (const float*)d_in[6];

    float* out_att = (float*)d_out;                          // [B,S,H]
    float* out_rel = out_att + (size_t)B_ * S_ * H_;         // [B,S,S]

    const size_t projElems = (size_t)B_ * S_ * H_;           // 16.78M
    char* ws = (char*)d_ws;
    __bf16* phi    = (__bf16*)(ws);                          // [B*S, H]
    __bf16* theta  = (__bf16*)(ws + projElems * 2);          // [B*S, H]
    __bf16* gT     = (__bf16*)(ws + projElems * 4);          // [B, H, S]
    float*  rowsum = (float*)(ws + projElems * 6);           // B*S floats
    float*  colsum = rowsum + (size_t)B_ * S_;               // B*S floats

    const float scale = 1.0f / sqrtf((float)D_);
    dim3 blk(256);

    // 1) phi/theta projections + transposed g (bf16 WMMA)
    proj3_kernel<<<dim3((B_ * S_) / BM, H_ / BN, 3), blk, 0, stream>>>(
        audio, Wphi, bphi, Wth, bth, Wg, bg, phi, theta, gT);

    // 2) R = relu(scale * phi @ theta^T) -> rel region of d_out (raw)
    rel_raw_kernel<<<dim3(S_ / BM, S_ / BN, B_), blk, 0, stream>>>(
        phi, theta, out_rel, scale);

    // 3/4) deterministic row & column sums of raw R
    rowsum_kernel<<<dim3(B_ * S_), blk, 0, stream>>>(out_rel, rowsum);
    colsum_kernel<<<dim3(S_ / 256, B_), blk, 0, stream>>>(out_rel, colsum);

    // 5) in-place symmetrize + row-normalize
    symnorm_kernel<<<dim3(S_ / TS, S_ / TS, B_), blk, 0, stream>>>(
        out_rel, rowsum, colsum);

    // 6) a_att = rel @ g + audio
    attn_kernel<<<dim3(S_ / BM, H_ / BN, B_), blk, 0, stream>>>(
        out_rel, gT, audio, out_att);
}
